// YOLACTDecoder_1176821040073
// MI455X (gfx1250) — matrix operand order repaired and checked by reading, not compiled, not run on hardware
//
#include <hip/hip_runtime.h>
#include <hip/hip_bf16.h>
#include <math.h>

typedef __attribute__((ext_vector_type(16))) _Float16 v16h;
typedef __attribute__((ext_vector_type(8)))  float    v8f;

#define A_ANCH 18525
#define NCLS   81
#define NFG    80
#define TOPN   200
#define MAXOBJ 100
#define PROTO  136
#define MROWS  (PROTO*PROTO)   // 18496
#define MTILES (MROWS/16)      // 1156
#define NTILES 7
#define NPAD   112             // 7 N-tiles of 16
#define ORIGIN 544
#define BATCH  2
#define MIN_SCORE 0.05f
#define NMS_T  0.5f

__device__ __forceinline__ float sigmoidf(float v) { return 1.f / (1.f + expf(-v)); }

// ---------------------------------------------------------------------------
// Kernel 1: per-anchor softmax + score transpose + box decode.
//   scoresT layout: [b][class][anchor]  (masked: -1 where anchor invalid)
//   boxes layout:   [b][anchor][4]      (x1,y1,x2,y2, clipped to [0,1])
// anchor_valid = max fg prob > 0.05  <=>  exp(max_fg_logit - mx)/sum > 0.05
// ---------------------------------------------------------------------------
__global__ void k_softmax_decode(const float* __restrict__ cls,
                                 const float* __restrict__ boxd,
                                 const float* __restrict__ anch,
                                 float* __restrict__ scoresT,
                                 float* __restrict__ boxes) {
  int idx = blockIdx.x * blockDim.x + threadIdx.x;
  if (idx >= BATCH * A_ANCH) return;
  int b = idx / A_ANCH, a = idx % A_ANCH;
  const float* l = cls + (size_t)idx * NCLS;
  float mx = l[0], mxfg = l[1];
  for (int c = 1; c < NCLS; c++) mxfg = fmaxf(mxfg, l[c]);
  mx = fmaxf(mx, mxfg);
  float sum = 0.f;
  for (int c = 0; c < NCLS; c++) sum += expf(l[c] - mx);
  float inv = 1.f / sum;
  bool valid = (expf(mxfg - mx) * inv) > MIN_SCORE;
  for (int c = 1; c < NCLS; c++) {
    float p = expf(l[c] - mx) * inv;
    scoresT[((size_t)(b * NFG + (c - 1))) * A_ANCH + a] = valid ? p : -1.0f;
  }
  float ax = anch[a*4+0], ay = anch[a*4+1], aw = anch[a*4+2], ah = anch[a*4+3];
  const float* d = boxd + (size_t)idx * 4;
  float cx = ax + d[0] * 0.1f * aw;
  float cy = ay + d[1] * 0.1f * ah;
  float w  = aw * expf(d[2] * 0.2f);
  float h  = ah * expf(d[3] * 0.2f);
  float x1 = cx - 0.5f * w, y1 = cy - 0.5f * h;
  boxes[(size_t)idx*4+0] = fminf(fmaxf(x1,     0.f), 1.f);
  boxes[(size_t)idx*4+1] = fminf(fmaxf(y1,     0.f), 1.f);
  boxes[(size_t)idx*4+2] = fminf(fmaxf(x1 + w, 0.f), 1.f);
  boxes[(size_t)idx*4+3] = fminf(fmaxf(y1 + h, 0.f), 1.f);
}

// ---------------------------------------------------------------------------
// Kernel 2: per-(batch,class) top-200 via 200 block-wide argmax passes over an
// f16 LDS copy of the 18525 scores (37 KB of LDS, zero re-reads from memory).
// ---------------------------------------------------------------------------
__global__ void k_topk(const float* __restrict__ scoresT,
                       int* __restrict__ order) {
  __shared__ _Float16 ssc[A_ANCH];
  __shared__ float rv[256];
  __shared__ int   ri[256];
  int bc = blockIdx.x;                     // b*NFG + c
  int tid = threadIdx.x;
  const float* src = scoresT + (size_t)bc * A_ANCH;
  for (int a = tid; a < A_ANCH; a += 256) ssc[a] = (_Float16)src[a];
  __syncthreads();
  for (int it = 0; it < TOPN; it++) {
    float best = -1e30f; int bi = -1;
    for (int a = tid; a < A_ANCH; a += 256) {
      float v = (float)ssc[a];
      if (v > best) { best = v; bi = a; }
    }
    rv[tid] = best; ri[tid] = bi;
    __syncthreads();
    for (int s = 128; s > 0; s >>= 1) {
      if (tid < s) {
        float vo = rv[tid + s]; int io = ri[tid + s];
        if (io >= 0 && (ri[tid] < 0 || vo > rv[tid] ||
                        (vo == rv[tid] && io < ri[tid]))) {
          rv[tid] = vo; ri[tid] = io;
        }
      }
      __syncthreads();
    }
    if (tid == 0) {
      int wsel = (ri[0] >= 0) ? ri[0] : 0;
      order[(size_t)bc * TOPN + it] = wsel;
      ssc[wsel] = (_Float16)(-2.0f);
    }
    __syncthreads();
  }
}

// ---------------------------------------------------------------------------
// Kernel 3: fast-NMS per (batch,class).  det j is dropped if any higher-rank
// valid det i<j has IoU > 0.5.  Output: keepsc[b][c][j] = score or -inf.
// ---------------------------------------------------------------------------
__global__ void k_nms(const float* __restrict__ scoresT,
                      const float* __restrict__ boxes,
                      const int* __restrict__ order,
                      float* __restrict__ keepsc) {
  __shared__ float bx[TOPN][4];
  __shared__ float sc[TOPN];
  int bc = blockIdx.x;
  int b = bc / NFG;
  int j = threadIdx.x;
  if (j < TOPN) {
    int anchor = order[(size_t)bc * TOPN + j];
    sc[j] = scoresT[(size_t)bc * A_ANCH + anchor];
    const float* bp = boxes + (size_t)(b * A_ANCH + anchor) * 4;
    bx[j][0] = bp[0]; bx[j][1] = bp[1]; bx[j][2] = bp[2]; bx[j][3] = bp[3];
  }
  __syncthreads();
  if (j >= TOPN) return;
  float x1j = bx[j][0], y1j = bx[j][1], x2j = bx[j][2], y2j = bx[j][3];
  float areaj = (x2j - x1j) * (y2j - y1j);
  float maxiou = 0.f;
  for (int i = 0; i < j; i++) {
    if (sc[i] <= -0.5f) continue;        // invalid dets cannot suppress
    float ix1 = fmaxf(bx[i][0], x1j), iy1 = fmaxf(bx[i][1], y1j);
    float ix2 = fminf(bx[i][2], x2j), iy2 = fminf(bx[i][3], y2j);
    float iw = fmaxf(ix2 - ix1, 0.f), ih = fmaxf(iy2 - iy1, 0.f);
    float inter = iw * ih;
    float areai = (bx[i][2] - bx[i][0]) * (bx[i][3] - bx[i][1]);
    float iou = inter / (areai + areaj - inter);
    maxiou = fmaxf(maxiou, iou);
  }
  bool keep = (maxiou <= NMS_T) && (sc[j] > -0.5f);
  keepsc[(size_t)bc * TOPN + j] = keep ? sc[j] : -__builtin_inff();
}

// ---------------------------------------------------------------------------
// Kernel 4: global top-100 per batch + gather of selected boxes/coefs.
// Writes labels/scores/valid directly to d_out tail; packs coefficients
// into the WMMA B layout coefB[b][k=0..31][n=0..111] (zero pad n>=100).
// ---------------------------------------------------------------------------
__global__ void k_select(const float* __restrict__ boxes,
                         const float* __restrict__ coef,
                         const int* __restrict__ order,
                         float* __restrict__ keepsc,
                         float* __restrict__ selbox,
                         float* __restrict__ coefB,
                         float* __restrict__ validf,
                         float* __restrict__ out_lab,
                         float* __restrict__ out_sc,
                         float* __restrict__ out_va) {
  __shared__ float rv[256];
  __shared__ int   ri[256];
  __shared__ int   o2[MAXOBJ];
  __shared__ float v2[MAXOBJ];
  int b = blockIdx.x, tid = threadIdx.x;
  const int FLATN = NFG * TOPN;            // 16000
  float* fk = keepsc + (size_t)b * FLATN;
  for (int it = 0; it < MAXOBJ; it++) {
    float best = -__builtin_inff(); int bi = -1;
    for (int f = tid; f < FLATN; f += 256) {
      float v = fk[f];
      if (v > best) { best = v; bi = f; }
    }
    if (bi < 0) bi = tid;                  // all remaining -inf: any index
    rv[tid] = best; ri[tid] = bi;
    __syncthreads();
    for (int s = 128; s > 0; s >>= 1) {
      if (tid < s) {
        float vo = rv[tid + s]; int io = ri[tid + s];
        if (vo > rv[tid] || (vo == rv[tid] && io < ri[tid])) {
          rv[tid] = vo; ri[tid] = io;
        }
      }
      __syncthreads();
    }
    if (tid == 0) { o2[it] = ri[0]; v2[it] = rv[0]; fk[ri[0]] = -__builtin_inff(); }
    __syncthreads();
  }
  if (tid < MAXOBJ) {
    int k = tid;
    int f = o2[k];
    float val = v2[k];
    bool valid = (val > -1e37f);
    int c = f / TOPN, j = f % TOPN;
    int anchor = order[(size_t)(b * NFG + c) * TOPN + j];
    for (int i = 0; i < 4; i++)
      selbox[(size_t)(b * MAXOBJ + k) * 4 + i] =
          valid ? boxes[(size_t)(b * A_ANCH + anchor) * 4 + i] : 0.f;
    for (int kk = 0; kk < 32; kk++)
      coefB[((size_t)b * 32 + kk) * NPAD + k] =
          valid ? coef[((size_t)(b * A_ANCH + anchor)) * 32 + kk] : 0.f;
    out_lab[b * MAXOBJ + k] = valid ? (float)c : -1.f;
    out_sc [b * MAXOBJ + k] = valid ? val : 0.f;
    out_va [b * MAXOBJ + k] = valid ? 1.f : 0.f;
    validf [b * MAXOBJ + k] = valid ? 1.f : 0.f;
  }
  // zero the pad columns n = 100..111
  for (int p = tid; p < 32 * (NPAD - MAXOBJ); p += 256) {
    int kk = p / (NPAD - MAXOBJ), n = MAXOBJ + p % (NPAD - MAXOBJ);
    coefB[((size_t)b * 32 + kk) * NPAD + n] = 0.f;
  }
}

// ---------------------------------------------------------------------------
// Kernel 5: proto-mask matmul via WMMA.  One wave32 per 16-row M-tile of
// m = proto[18496 x 32] @ coefB[32 x 112]; the wave loads its A tile ONCE
// into registers and loops over all 7 N-tiles (7 x v_wmma_f32_16x16x32_f16,
// A traffic cut 7x vs tile-per-wave).  K=32 is exactly one WMMA per tile.
// Fused epilogue: sigmoid + crop at proto resolution.  The guarded epilogue
// re-converges before the next WMMA, so EXEC is all-1s at every WMMA.
// ---------------------------------------------------------------------------
__global__ void k_wmma_mask(const float* __restrict__ proto,
                            const float* __restrict__ coefB,
                            const float* __restrict__ selbox,
                            float* __restrict__ maskp) {
  int lane = threadIdx.x & 31;
  int wid  = blockIdx.x * (blockDim.x >> 5) + (threadIdx.x >> 5);
  int mtile = wid % MTILES;
  int b     = wid / MTILES;
  int hh  = lane >> 4;          // half-wave: 0 or 1
  int r16 = lane & 15;
  int m = mtile * 16 + r16;     // A-matrix row for this lane
  const float* ap = proto + ((size_t)(b * MROWS + m)) * 32;
  const float* bp = coefB + (size_t)b * 32 * NPAD;

  // A tile, loaded once: 16-bit A 16x32 layout,
  // VGPR0-3 -> K = hh*8 + 2i(+1); VGPR4-7 -> K = 16 + hh*8 + 2(i-4)(+1)
  v16h aV;
#pragma unroll
  for (int i = 0; i < 8; i++) {
    int ka = (i < 4) ? (hh * 8 + 2 * i) : (16 + hh * 8 + 2 * (i - 4));
    aV[2*i]   = (_Float16)ap[ka];
    aV[2*i+1] = (_Float16)ap[ka + 1];
  }

#pragma unroll
  for (int ntile = 0; ntile < NTILES; ntile++) {
    int n = ntile * 16 + r16;   // B/D column for this lane
    // B 32x16 layout: same K order as A (VGPR0-3 -> K = hh*8 + 2i(+1); VGPR4-7 -> K = 16 + hh*8 + 2(i-4)(+1))
    v16h bV;
#pragma unroll
    for (int i = 0; i < 8; i++) {
      int kb = (i < 4) ? (hh * 8 + 2 * i) : (16 + hh * 8 + 2 * (i - 4));
      bV[2*i]   = (_Float16)bp[(size_t)kb       * NPAD + n];
      bV[2*i+1] = (_Float16)bp[(size_t)(kb + 1) * NPAD + n];
    }
    v8f acc = {};
    acc = __builtin_amdgcn_wmma_f32_16x16x32_f16(
        /*neg_a=*/false, aV, /*neg_b=*/false, bV,
        /*c_mod=*/(short)0, acc, /*reuse_a=*/false, /*reuse_b=*/false);
    if (n < MAXOBJ) {           // pad columns skip epilogue (re-converges)
      const float* bb = selbox + (size_t)(b * MAXOBJ + n) * 4;
      float bx1 = bb[0], by1 = bb[1], bx2 = bb[2], by2 = bb[3];
      float cx1 = fmaxf(fminf(bx1, bx2) * (float)PROTO - 1.f, 0.f);
      float cx2 = fminf(fmaxf(bx1, bx2) * (float)PROTO + 1.f, (float)PROTO);
      float cy1 = fmaxf(fminf(by1, by2) * (float)PROTO - 1.f, 0.f);
      float cy2 = fminf(fmaxf(by1, by2) * (float)PROTO + 1.f, (float)PROTO);
#pragma unroll
      for (int r = 0; r < 8; r++) {
        int md = mtile * 16 + r + 8 * hh;  // D layout: VGPR r -> M=r (+8 upper)
        int y = md / PROTO, x = md % PROTO;
        bool keep = ((float)x >= cx1) && ((float)x < cx2) &&
                    ((float)y >= cy1) && ((float)y < cy2);
        float s = sigmoidf(acc[r]);
        maskp[((size_t)(b * MAXOBJ + n) * PROTO + y) * PROTO + x] = keep ? s : 0.f;
      }
    }
  }
}

// ---------------------------------------------------------------------------
// Kernel 6: bilinear upsample 136 -> 544 (half-pixel centers, scale 0.25),
// threshold at 0.5, gate by validity.  Bandwidth-dominant pass (~237 MB
// written): each thread produces 4 consecutive X pixels and stores a float4
// (global_store_b128), sharing the two source rows across the quad.
// ---------------------------------------------------------------------------
__global__ void k_upsample(const float* __restrict__ maskp,
                           const float* __restrict__ validf,
                           float* __restrict__ outm) {
  size_t gid = (size_t)blockIdx.x * blockDim.x + threadIdx.x;  // 1 per 4 px
  const size_t total4 = (size_t)BATCH * MAXOBJ * ORIGIN * (ORIGIN / 4);
  if (gid >= total4) return;
  int X4 = (int)(gid % (ORIGIN / 4)) * 4;
  int Y  = (int)((gid / (ORIGIN / 4)) % ORIGIN);
  int n  = (int)((gid / ((size_t)(ORIGIN / 4) * ORIGIN)) % MAXOBJ);
  int b  = (int)(gid / ((size_t)(ORIGIN / 4) * ORIGIN * MAXOBJ));
  float4 res = make_float4(0.f, 0.f, 0.f, 0.f);
  if (validf[b * MAXOBJ + n] > 0.5f) {
    float sy = (Y + 0.5f) * 0.25f - 0.5f;
    float fy0 = floorf(sy);
    float ty = sy - fy0;
    int y0 = min(max((int)fy0,     0), PROTO - 1);
    int y1 = min(max((int)fy0 + 1, 0), PROTO - 1);
    const float* base = maskp + (size_t)(b * MAXOBJ + n) * PROTO * PROTO;
    const float* row0 = base + (size_t)y0 * PROTO;
    const float* row1 = base + (size_t)y1 * PROTO;
    float o[4];
#pragma unroll
    for (int i = 0; i < 4; i++) {
      float sx = (X4 + i + 0.5f) * 0.25f - 0.5f;
      float fx0 = floorf(sx);
      float tx = sx - fx0;
      int x0 = min(max((int)fx0,     0), PROTO - 1);
      int x1 = min(max((int)fx0 + 1, 0), PROTO - 1);
      float v0 = row0[x0] * (1.f - tx) + row0[x1] * tx;
      float v1 = row1[x0] * (1.f - tx) + row1[x1] * tx;
      float mv = v0 * (1.f - ty) + v1 * ty;
      o[i] = (mv > 0.5f) ? 1.f : 0.f;
    }
    res = make_float4(o[0], o[1], o[2], o[3]);
  }
  ((float4*)outm)[gid] = res;
}

// ---------------------------------------------------------------------------
extern "C" void kernel_launch(void* const* d_in, const int* in_sizes, int n_in,
                              void* d_out, int out_size, void* d_ws, size_t ws_size,
                              hipStream_t stream) {
  const float* cls   = (const float*)d_in[0];   // [2, 18525, 81]
  const float* boxd  = (const float*)d_in[1];   // [2, 18525, 4]
  const float* coef  = (const float*)d_in[2];   // [2, 18525, 32]
  const float* proto = (const float*)d_in[3];   // [2, 136, 136, 32]
  const float* anch  = (const float*)d_in[4];   // [18525, 4]
  float* out = (float*)d_out;

  // workspace carve-up (all L2-resident: ~28 MB total)
  float* scoresT = (float*)d_ws;                                     // 2*80*18525
  float* boxes   = scoresT + (size_t)BATCH * NFG * A_ANCH;           // 2*18525*4
  int*   order   = (int*)(boxes + (size_t)BATCH * A_ANCH * 4);       // 2*80*200
  float* keepsc  = (float*)(order + (size_t)BATCH * NFG * TOPN);     // 2*80*200
  float* selbox  = keepsc + (size_t)BATCH * NFG * TOPN;              // 2*100*4
  float* coefB   = selbox + (size_t)BATCH * MAXOBJ * 4;              // 2*32*112
  float* validf  = coefB + (size_t)BATCH * 32 * NPAD;                // 2*100
  float* maskp   = validf + (size_t)BATCH * MAXOBJ;                  // 2*100*136*136

  float* out_lab = out + (size_t)BATCH * MAXOBJ * ORIGIN * ORIGIN;
  float* out_sc  = out_lab + BATCH * MAXOBJ;
  float* out_va  = out_sc + BATCH * MAXOBJ;

  // 1) softmax + decode
  {
    int threads = BATCH * A_ANCH;
    k_softmax_decode<<<(threads + 255) / 256, 256, 0, stream>>>(
        cls, boxd, anch, scoresT, boxes);
  }
  // 2) per-class top-200
  k_topk<<<BATCH * NFG, 256, 0, stream>>>(scoresT, order);
  // 3) fast NMS
  k_nms<<<BATCH * NFG, 256, 0, stream>>>(scoresT, boxes, order, keepsc);
  // 4) global top-100 + gather
  k_select<<<BATCH, 256, 0, stream>>>(boxes, coef, order, keepsc, selbox,
                                      coefB, validf, out_lab, out_sc, out_va);
  // 5) WMMA proto-mask matmul + sigmoid + crop
  //    one wave per M-tile, 7 WMMAs per wave (A reused across all N-tiles)
  {
    int waves = BATCH * MTILES;             // 2312 waves
    k_wmma_mask<<<waves / 4, 128, 0, stream>>>(proto, coefB, selbox, maskp);
  }
  // 6) bilinear upsample + threshold (bandwidth-dominant: ~237 MB write)
  {
    size_t total4 = (size_t)BATCH * MAXOBJ * ORIGIN * (ORIGIN / 4);
    k_upsample<<<(unsigned)((total4 + 255) / 256), 256, 0, stream>>>(
        maskp, validf, out);
  }
}
